// DragonGCN_48146583388505
// MI455X (gfx1250) — compile-verified, matrix operand order checked by reading, not run on hardware
//
#include <hip/hip_runtime.h>
#include <hip/hip_bf16.h>

// ---------------------------------------------------------------------------
// DragonGCN for MI455X (gfx1250, wave32):
//   item MLP encoder  -> v_wmma_f32_16x16x32_bf16, 64-row macro-tile,
//                        1 b128 load per wmma (A amortized over 4 n-tiles,
//                        B amortized over 4 m-subtiles)
//   LightGCN 2-hop    -> edge-parallel SpMM, L2-resident state (38.4MB << 192MB)
// ---------------------------------------------------------------------------

typedef __attribute__((ext_vector_type(16))) __bf16 v16bf;
typedef __attribute__((ext_vector_type(8)))  __bf16 v8bf;
typedef __attribute__((ext_vector_type(8)))  float  v8f;

union Frag { v16bf v; v8bf h[2]; };

constexpr int NUM_USER  = 100000;
constexpr int NUM_ITEM  = 50000;
constexpr int NUM_NODES = NUM_USER + NUM_ITEM;
constexpr int FEAT      = 512;
constexpr int HID       = 256;   // 4 * DIM_LATENT
constexpr int DL        = 64;    // DIM_LATENT
constexpr int MT        = 64;    // item rows per block (4 x 16-row subtiles)
constexpr float NEG_SLOPE = 0.01f;

static __device__ __forceinline__ __bf16 f2bf(float f) {
  // round-to-nearest-even f32 -> bf16
  unsigned u = __builtin_bit_cast(unsigned, f);
  unsigned r = (u + 0x7FFFu + ((u >> 16) & 1u)) >> 16;
  return __builtin_bit_cast(__bf16, (unsigned short)r);
}

// ---------------------------------------------------------------------------
// conversion kernels (run once per launch, trivially memory-bound)
// ---------------------------------------------------------------------------
__global__ void cvt_bf16_kernel(const float* __restrict__ in, __bf16* __restrict__ out,
                                long long n8) {    // n8 = n/8
  long long i = (long long)blockIdx.x * blockDim.x + threadIdx.x;
  if (i >= n8) return;
  const float4 f0 = ((const float4*)in)[2 * i];
  const float4 f1 = ((const float4*)in)[2 * i + 1];
  v8bf o;
  o[0] = f2bf(f0.x); o[1] = f2bf(f0.y); o[2] = f2bf(f0.z); o[3] = f2bf(f0.w);
  o[4] = f2bf(f1.x); o[5] = f2bf(f1.y); o[6] = f2bf(f1.z); o[7] = f2bf(f1.w);
  ((v8bf*)out)[i] = o;
}

// out[n*K + k] = bf16(in[k*N + n])   (transpose K-major -> N-major)
__global__ void tcvt_bf16_kernel(const float* __restrict__ in, __bf16* __restrict__ out,
                                 int K, int N) {
  int idx = blockIdx.x * blockDim.x + threadIdx.x;
  if (idx >= K * N) return;
  int n = idx / K;
  int k = idx % K;
  out[idx] = f2bf(in[(long long)k * N + n]);
}

// ---------------------------------------------------------------------------
// utility kernels
// ---------------------------------------------------------------------------
__global__ void zero_u32_kernel(unsigned* __restrict__ p, long long n) {
  long long i = (long long)blockIdx.x * blockDim.x + threadIdx.x;
  if (i < n) p[i] = 0u;
}

__global__ void degree_kernel(const int* __restrict__ src, unsigned* __restrict__ deg, int E) {
  int e = blockIdx.x * blockDim.x + threadIdx.x;
  if (e < E) atomicAdd(&deg[src[e]], 1u);
}

__global__ void dinv_kernel(const unsigned* __restrict__ deg, float* __restrict__ dinv, int n) {
  int i = blockIdx.x * blockDim.x + threadIdx.x;
  if (i < n) {
    unsigned d = deg[i];
    dinv[i] = (d > 0u) ? rsqrtf((float)d) : 0.0f;
  }
}

__global__ void copy_f32_kernel(const float* __restrict__ in, float* __restrict__ out, long long n) {
  long long i = (long long)blockIdx.x * blockDim.x + threadIdx.x;
  if (i < n) out[i] = in[i];
}

__global__ void add_f32_kernel(const float* __restrict__ in, float* __restrict__ out, long long n) {
  long long i = (long long)blockIdx.x * blockDim.x + threadIdx.x;
  if (i < n) out[i] += in[i];
}

__global__ void scale_f32_kernel(float* __restrict__ out, float s, long long n) {
  long long i = (long long)blockIdx.x * blockDim.x + threadIdx.x;
  if (i < n) out[i] *= s;
}

// ---------------------------------------------------------------------------
// Fused item encoder, 64-row macro-tile, 128 threads = 4 waves.
// GEMM1 (per wave): 4 m-subtiles x 4 n-tiles accumulators; per k-step issue
//   4 A-frag + 4 B-frag loads (16 x b128) feeding 16 wmma  -> 1 load / wmma.
// GEMM2: hidden (64x256 bf16, 32 KB LDS) @ W2t -> 64x64; one B-frag per
//   k-step feeds 4 wmma; A-frags via ds_load_b128.
// Fragment layout per ISA 7.12.2 (16-bit A 16x32): lane l15 = row, K runs
// {k0+half*8 .. +7} and {k0+16+half*8 .. +7} -> two 16B contiguous loads.
// ---------------------------------------------------------------------------
__global__ __launch_bounds__(128) void
mlp_wmma_kernel(const __bf16* __restrict__ featbf,
                const __bf16* __restrict__ W1t,  // [HID][FEAT] bf16
                const float*  __restrict__ b1,
                const __bf16* __restrict__ W2t,  // [DL][HID] bf16
                const float*  __restrict__ b2,
                float* __restrict__ x_out) {
  __shared__ __bf16 hidden[MT][HID];   // 32 KB of 320 KB/WGP LDS

  const int m0   = blockIdx.x * MT;
  const int wave = threadIdx.x >> 5;   // 0..3
  const int lane = threadIdx.x & 31;
  const int half = lane >> 4;
  const int l15  = lane & 15;

  // ---- GEMM1 --------------------------------------------------------------
  // clamped A-row pointers (phantom rows computed but never stored)
  const __bf16* arow[4];
#pragma unroll
  for (int ms = 0; ms < 4; ++ms) {
    int r = m0 + ms * 16 + l15;
    r = (r < NUM_ITEM) ? r : (NUM_ITEM - 1);
    arow[ms] = featbf + (long long)r * FEAT;
  }
  const int nbase = wave * 64;         // this wave's 4 n-tiles

  v8f cacc[4][4] = {};                 // [m-subtile][n-tile], 128 VGPRs
  for (int k0 = 0; k0 < FEAT; k0 += 32) {
    Frag a[4];
#pragma unroll
    for (int ms = 0; ms < 4; ++ms) {
      a[ms].h[0] = *(const v8bf*)(arow[ms] + k0 + half * 8);
      a[ms].h[1] = *(const v8bf*)(arow[ms] + k0 + 16 + half * 8);
    }
    if (k0 + 32 < FEAT)
      __builtin_prefetch(arow[0] + k0 + 32, 0, 1);   // global_prefetch_b8
#pragma unroll
    for (int t = 0; t < 4; ++t) {
      const __bf16* brow = W1t + (long long)(nbase + t * 16 + l15) * FEAT;
      Frag b;
      b.h[0] = *(const v8bf*)(brow + k0 + half * 8);
      b.h[1] = *(const v8bf*)(brow + k0 + 16 + half * 8);
#pragma unroll
      for (int ms = 0; ms < 4; ++ms) {
        cacc[ms][t] = __builtin_amdgcn_wmma_f32_16x16x32_bf16(
                          /*neg_a=*/false, a[ms].v, /*neg_b=*/false, b.v,
                          /*c_mod=*/(short)0, cacc[ms][t],
                          /*reuse_a=*/false, /*reuse_b=*/false);
      }
    }
  }

  // bias + LeakyReLU -> LDS (C/D layout: M = r + 8*half, N = l15)
#pragma unroll
  for (int t = 0; t < 4; ++t) {
    const int n = nbase + t * 16 + l15;
    const float bias = b1[n];
#pragma unroll
    for (int ms = 0; ms < 4; ++ms) {
#pragma unroll
      for (int r = 0; r < 8; ++r) {
        float v = cacc[ms][t][r] + bias;
        v = (v > 0.0f) ? v : NEG_SLOPE * v;
        hidden[ms * 16 + r + half * 8][n] = f2bf(v);
      }
    }
  }
  __syncthreads();

  // ---- GEMM2: one 16-col tile per wave, 4 m-subtiles ----------------------
  const __bf16* brow2 = W2t + (long long)(wave * 16 + l15) * HID;
  v8f c2[4] = {};
  for (int k0 = 0; k0 < HID; k0 += 32) {
    Frag b;
    b.h[0] = *(const v8bf*)(brow2 + k0 + half * 8);
    b.h[1] = *(const v8bf*)(brow2 + k0 + 16 + half * 8);
#pragma unroll
    for (int ms = 0; ms < 4; ++ms) {
      const __bf16* hrow = &hidden[ms * 16 + l15][0];
      Frag a;
      a.h[0] = *(const v8bf*)(hrow + k0 + half * 8);        // ds_load_b128
      a.h[1] = *(const v8bf*)(hrow + k0 + 16 + half * 8);
      c2[ms] = __builtin_amdgcn_wmma_f32_16x16x32_bf16(
                   false, a.v, false, b.v, (short)0, c2[ms], false, false);
    }
  }
  {
    const int n = wave * 16 + l15;
    const float bias = b2[n];
#pragma unroll
    for (int ms = 0; ms < 4; ++ms) {
#pragma unroll
      for (int r = 0; r < 8; ++r) {
        const int m = m0 + ms * 16 + r + half * 8;   // item row
        if (m < NUM_ITEM)
          x_out[(long long)(NUM_USER + m) * DL + n] = c2[ms][r] + bias;
      }
    }
  }
}

// ---------------------------------------------------------------------------
// Edge-parallel SpMM hop: x_out[src] += dinv[src]*dinv[dst] * x_in[dst]
// One thread per (edge, 4-dim group): float4 gather (L2-resident) + 4 native
// f32 atomics (relaxed, agent scope -> global_atomic_add_f32).
// ---------------------------------------------------------------------------
__global__ void spmm_kernel(const int* __restrict__ src, const int* __restrict__ dst,
                            const float* __restrict__ dinv,
                            const float* __restrict__ x_in, float* __restrict__ x_out,
                            int E) {
  long long gid = (long long)blockIdx.x * blockDim.x + threadIdx.x;
  long long total = (long long)E * (DL / 4);
  if (gid >= total) return;
  const int e = (int)(gid >> 4);          // DL/4 == 16 groups per edge
  const int g = ((int)gid & 15) * 4;
  const int s = src[e];
  const int d = dst[e];
  const float w = dinv[s] * dinv[d];
  const float4 v = *(const float4*)&x_in[(long long)d * DL + g];
  float* o = &x_out[(long long)s * DL + g];
  __hip_atomic_fetch_add(o + 0, w * v.x, __ATOMIC_RELAXED, __HIP_MEMORY_SCOPE_AGENT);
  __hip_atomic_fetch_add(o + 1, w * v.y, __ATOMIC_RELAXED, __HIP_MEMORY_SCOPE_AGENT);
  __hip_atomic_fetch_add(o + 2, w * v.z, __ATOMIC_RELAXED, __HIP_MEMORY_SCOPE_AGENT);
  __hip_atomic_fetch_add(o + 3, w * v.w, __ATOMIC_RELAXED, __HIP_MEMORY_SCOPE_AGENT);
}

// ---------------------------------------------------------------------------
// launch
// ---------------------------------------------------------------------------
static inline size_t alignup(size_t x) { return (x + 255) & ~(size_t)255; }

extern "C" void kernel_launch(void* const* d_in, const int* in_sizes, int n_in,
                              void* d_out, int out_size, void* d_ws, size_t ws_size,
                              hipStream_t stream) {
  const float* features = (const float*)d_in[0];
  const int*   edge     = (const int*)d_in[1];
  const float* pref     = (const float*)d_in[2];
  const float* W1       = (const float*)d_in[3];
  const float* b1       = (const float*)d_in[4];
  const float* W2       = (const float*)d_in[5];
  const float* b2       = (const float*)d_in[6];
  float*       acc      = (float*)d_out;        // [NUM_NODES, DL]

  const int E = in_sizes[1] / 2;                // directed edges
  const int* src = edge;
  const int* dst = edge + E;

  // workspace layout
  char* ws = (char*)d_ws;
  unsigned* deg    = (unsigned*)ws;  ws += alignup(sizeof(unsigned) * NUM_NODES);
  float*    dinv   = (float*)ws;     ws += alignup(sizeof(float) * NUM_NODES);
  float*    xa     = (float*)ws;     ws += alignup(sizeof(float) * (size_t)NUM_NODES * DL);
  float*    xb     = (float*)ws;     ws += alignup(sizeof(float) * (size_t)NUM_NODES * DL);
  __bf16*   featbf = (__bf16*)ws;    ws += alignup(sizeof(__bf16) * (size_t)NUM_ITEM * FEAT);
  __bf16*   w1t    = (__bf16*)ws;    ws += alignup(sizeof(__bf16) * (size_t)FEAT * HID);
  __bf16*   w2t    = (__bf16*)ws;

  const long long NODE_ELEMS = (long long)NUM_NODES * DL;
  const int TB = 256;
  auto blocks = [&](long long n) { return (unsigned)((n + TB - 1) / TB); };

  // 0) precision staging: f32 -> bf16 (features), transposed bf16 weights
  cvt_bf16_kernel<<<blocks((long long)NUM_ITEM * FEAT / 8), TB, 0, stream>>>(
      features, featbf, (long long)NUM_ITEM * FEAT / 8);
  tcvt_bf16_kernel<<<blocks(FEAT * HID), TB, 0, stream>>>(W1, w1t, FEAT, HID);
  tcvt_bf16_kernel<<<blocks(HID * DL), TB, 0, stream>>>(W2, w2t, HID, DL);

  // 1) degrees -> d^-1/2
  zero_u32_kernel<<<blocks(NUM_NODES), TB, 0, stream>>>(deg, NUM_NODES);
  degree_kernel<<<blocks(E), TB, 0, stream>>>(src, deg, E);
  dinv_kernel<<<blocks(NUM_NODES), TB, 0, stream>>>(deg, dinv, NUM_NODES);

  // 2) item encoder (WMMA) -> xa item rows; preference -> xa user rows
  mlp_wmma_kernel<<<(NUM_ITEM + MT - 1) / MT, 128, 0, stream>>>(
      featbf, w1t, b1, w2t, b2, xa);
  copy_f32_kernel<<<blocks((long long)NUM_USER * DL), TB, 0, stream>>>(
      pref, xa, (long long)NUM_USER * DL);

  // 3) acc = ego
  copy_f32_kernel<<<blocks(NODE_ELEMS), TB, 0, stream>>>(xa, acc, NODE_ELEMS);

  // 4) hop 1: xb = A_norm xa ; acc += xb
  zero_u32_kernel<<<blocks(NODE_ELEMS), TB, 0, stream>>>((unsigned*)xb, NODE_ELEMS);
  spmm_kernel<<<blocks((long long)E * (DL / 4)), TB, 0, stream>>>(src, dst, dinv, xa, xb, E);
  add_f32_kernel<<<blocks(NODE_ELEMS), TB, 0, stream>>>(xb, acc, NODE_ELEMS);

  // 5) hop 2: xa = A_norm xb ; acc += xa
  zero_u32_kernel<<<blocks(NODE_ELEMS), TB, 0, stream>>>((unsigned*)xa, NODE_ELEMS);
  spmm_kernel<<<blocks((long long)E * (DL / 4)), TB, 0, stream>>>(src, dst, dinv, xb, xa, E);
  add_f32_kernel<<<blocks(NODE_ELEMS), TB, 0, stream>>>(xa, acc, NODE_ELEMS);

  // 6) mean over [x0, x1, x2]
  scale_f32_kernel<<<blocks(NODE_ELEMS), TB, 0, stream>>>(acc, 1.0f / 3.0f, NODE_ELEMS);
}
